// SelfAttentionHead_36103495090252
// MI455X (gfx1250) — matrix operand order repaired and checked by reading, not compiled, not run on hardware
//
#include <hip/hip_runtime.h>
#include <hip/hip_bf16.h>

typedef __attribute__((ext_vector_type(16))) _Float16 v16h;
typedef __attribute__((ext_vector_type(8)))  _Float16 v8h;
typedef __attribute__((ext_vector_type(8)))  float    v8f;
typedef int b128_t __attribute__((__vector_size__(16)));

#define T_DIM 4096
#define E_DIM 2048
#define D_DIM 256

#if __has_builtin(__builtin_amdgcn_global_load_async_to_lds_b128)
#define HAVE_ASYNC_LDS 1
#endif

__device__ __forceinline__ v8f wmma_f16(v16h a, v16h b, v8f c) {
    // D = A(16x32 f16) * B(32x16 f16) + C(16x16 f32)
    return __builtin_amdgcn_wmma_f32_16x16x32_f16(
        false, a, false, b, (short)0, c, false, false);
}

// 16-byte global -> LDS copy; async path (ASYNCcnt) when available.
__device__ __forceinline__ void copy16_g2l(const _Float16* g, _Float16* l) {
#ifdef HAVE_ASYNC_LDS
    __builtin_amdgcn_global_load_async_to_lds_b128(
        (__attribute__((address_space(1))) b128_t*)(unsigned long long)(uintptr_t)g,
        (__attribute__((address_space(3))) b128_t*)(unsigned int)(uintptr_t)l,
        0, 0);
#else
    *(v8h*)l = *(const v8h*)g;
#endif
}

__device__ __forceinline__ void wait_async_copies() {
#ifdef HAVE_ASYNC_LDS
#if __has_builtin(__builtin_amdgcn_s_wait_asynccnt)
    __builtin_amdgcn_s_wait_asynccnt(0);
#else
    asm volatile("s_wait_asynccnt 0" ::: "memory");
#endif
#endif
}

__device__ __forceinline__ float fast_rcp(float v) {
#if __has_builtin(__builtin_amdgcn_rcpf)
    return __builtin_amdgcn_rcpf(v);
#else
    return 1.0f / v;
#endif
}

__device__ __forceinline__ float grp16_max(float v) {
    v = fmaxf(v, __shfl_xor(v, 1, 32));
    v = fmaxf(v, __shfl_xor(v, 2, 32));
    v = fmaxf(v, __shfl_xor(v, 4, 32));
    v = fmaxf(v, __shfl_xor(v, 8, 32));
    return v;
}
__device__ __forceinline__ float grp16_sum(float v) {
    v += __shfl_xor(v, 1, 32);
    v += __shfl_xor(v, 2, 32);
    v += __shfl_xor(v, 4, 32);
    v += __shfl_xor(v, 8, 32);
    return v;
}

// ---------------------------------------------------------------------------
// Kernel 0: one-time weight prep.  Wt[3][E][D] f16 = transpose+convert.
// ---------------------------------------------------------------------------
__global__ __launch_bounds__(256) void prep_w_kernel(
    const float* __restrict__ Wq, const float* __restrict__ Wk,
    const float* __restrict__ Wv, _Float16* __restrict__ Wt)
{
    const size_t i = (size_t)blockIdx.x * blockDim.x + threadIdx.x;
    const size_t per = (size_t)E_DIM * D_DIM;
    if (i >= 3 * per) return;
    const int    mat = (int)(i / per);
    const size_t rem = i % per;
    const int    e   = (int)(rem / D_DIM);
    const int    d   = (int)(rem % D_DIM);
    const float* W = (mat == 0) ? Wq : ((mat == 1) ? Wk : Wv);
    Wt[i] = (_Float16)W[(size_t)d * E_DIM + e];
}

// ---------------------------------------------------------------------------
// Kernel 1: Q/K/V projections.  One wave computes a 16x64 output tile.
// ---------------------------------------------------------------------------
__global__ __launch_bounds__(32) void proj_qkv_kernel(
    const float* __restrict__ x, const _Float16* __restrict__ Wt,
    _Float16* __restrict__ Qh, _Float16* __restrict__ Kt,
    _Float16* __restrict__ Vh)
{
    const int lane = threadIdx.x & 31;
    const int m16  = lane & 15;
    const int half = lane >> 4;
    const int row0 = blockIdx.x * 16;
    const int col0 = blockIdx.y * 64;
    const int mat  = blockIdx.z;
    const _Float16* Wm = Wt + (size_t)mat * E_DIM * D_DIM;

    v8f acc[4];
    #pragma unroll
    for (int t = 0; t < 4; ++t) acc[t] = (v8f){};

    for (int e0 = 0; e0 < E_DIM; e0 += 32) {
        const float* xr = x + (size_t)(row0 + m16) * E_DIM + e0;
        const float* xa = xr + 8 * half;
        const float* xb = xr + 16 + 8 * half;
        v16h a;
        #pragma unroll
        for (int i = 0; i < 8; ++i) {
            a[i]     = (_Float16)xa[i];
            a[8 + i] = (_Float16)xb[i];
        }
        #pragma unroll
        for (int t = 0; t < 4; ++t) {
            const _Float16* wp = Wm + (size_t)e0 * D_DIM + col0 + 16 * t + m16;
            v16h b;
            #pragma unroll
            for (int i = 0; i < 8; ++i) {
                b[i]     = wp[(size_t)(8 * half + i) * D_DIM];
                b[8 + i] = wp[(size_t)(16 + 8 * half + i) * D_DIM];
            }
            acc[t] = wmma_f16(a, b, acc[t]);
        }
    }

    if (mat == 0) {
        #pragma unroll
        for (int t = 0; t < 4; ++t)
            #pragma unroll
            for (int r = 0; r < 8; ++r)
                Qh[(size_t)(row0 + r + 8 * half) * D_DIM + col0 + 16 * t + m16] =
                    (_Float16)acc[t][r];
    } else if (mat == 1) {
        #pragma unroll
        for (int t = 0; t < 4; ++t)
            #pragma unroll
            for (int r = 0; r < 8; ++r)
                Kt[(size_t)(col0 + 16 * t + m16) * T_DIM + row0 + r + 8 * half] =
                    (_Float16)acc[t][r];
    } else {
        #pragma unroll
        for (int t = 0; t < 4; ++t)
            #pragma unroll
            for (int r = 0; r < 8; ++r)
                Vh[(size_t)(row0 + r + 8 * half) * D_DIM + col0 + 16 * t + m16] =
                    (_Float16)acc[t][r];
    }
}

// ---------------------------------------------------------------------------
// Kernel 2: causal flash attention, block-cooperative LDS staging.
// 1 block = 128 rows = 8 waves x 16 rows.  Per j-block, the 256 threads
// async-copy K (16KB) + V (16KB) slices into LDS once; all 8 waves consume
// them.  Online softmax in C-layout; LDS round-trip for C->A relayout of P.
// ---------------------------------------------------------------------------
#define KT_PITCH 40    // halves per ktile row (80B, 16B-aligned, bank-spread)
#define VT_PITCH 264   // halves per vtile row (528B, 16B-aligned, bank-spread)

__global__ __launch_bounds__(256) void flash_attn_kernel(
    const _Float16* __restrict__ Qh, const _Float16* __restrict__ Kt,
    const _Float16* __restrict__ Vh, float* __restrict__ out)
{
    __shared__ _Float16 ktile[256 * KT_PITCH];  // Kt[:, j0:j0+32)  (20.0 KB)
    __shared__ _Float16 vtile[32 * VT_PITCH];   // V[j0:j0+32, :]   (16.5 KB)
    __shared__ _Float16 pbuf[8][16 * 32];       // per-wave P tiles ( 8.0 KB)

    const int tid  = threadIdx.x;
    const int lane = tid & 31;
    const int wave = tid >> 5;
    const int m16  = lane & 15;
    const int half = lane >> 4;
    const int i0_blk = blockIdx.x * 128;
    const int i0   = i0_blk + wave * 16;        // this wave's first output row

    // Q tile 16x256 as 8 A-operands (registers, reused for every column block)
    v16h qa[8];
    #pragma unroll
    for (int c = 0; c < 8; ++c) {
        const _Float16* qr = Qh + (size_t)(i0 + m16) * D_DIM + 32 * c;
        v8h lo = *(const v8h*)(qr + 8 * half);
        v8h hi = *(const v8h*)(qr + 16 + 8 * half);
        qa[c] = __builtin_shufflevector(lo, hi, 0,1,2,3,4,5,6,7,
                                                 8,9,10,11,12,13,14,15);
    }

    float mrow[8], lrow[8];
    v8f o[16];
    #pragma unroll
    for (int r = 0; r < 8; ++r) { mrow[r] = -3e38f; lrow[r] = 0.f; }
    #pragma unroll
    for (int t = 0; t < 16; ++t) o[t] = (v8f){};

    const float scale = 0.0625f;   // 1/sqrt(256)

    for (int j0 = 0; j0 <= i0_blk + 127; j0 += 32) {
        // ---- cooperative async staging: K slice + V slice -> LDS ----
        {
            // ktile[d][0..31] = Kt[d][j0..j0+31], d = tid  (64B per thread)
            const _Float16* gk = Kt + (size_t)tid * T_DIM + j0;
            _Float16*       lk = ktile + tid * KT_PITCH;
            #pragma unroll
            for (int i = 0; i < 4; ++i) copy16_g2l(gk + 8 * i, lk + 8 * i);

            // vtile[j][0..255] = V[j0+j][:], 8 threads per row (coalesced)
            const int jr = tid >> 3, seg = tid & 7;
            const _Float16* gv = Vh + (size_t)(j0 + jr) * D_DIM + seg * 32;
            _Float16*       lv = vtile + jr * VT_PITCH + seg * 32;
            #pragma unroll
            for (int i = 0; i < 4; ++i) copy16_g2l(gv + 8 * i, lv + 8 * i);
        }
        wait_async_copies();
        __syncthreads();

        if (j0 <= i0 + 15) {   // wave-uniform: causal bound for this wave
            // ---- scores: two 16x16 tiles over columns [j0, j0+32) ----
            v8f s0 = {}, s1 = {};
            #pragma unroll
            for (int c = 0; c < 8; ++c) {
                const _Float16* kp = ktile + (32 * c) * KT_PITCH + m16;
                v16h b0, b1;
                #pragma unroll
                for (int i = 0; i < 8; ++i) {
                    b0[i]     = kp[(8 * half + i) * KT_PITCH];
                    b0[8 + i] = kp[(16 + 8 * half + i) * KT_PITCH];
                    b1[i]     = kp[(8 * half + i) * KT_PITCH + 16];
                    b1[8 + i] = kp[(16 + 8 * half + i) * KT_PITCH + 16];
                }
                s0 = wmma_f16(qa[c], b0, s0);
                s1 = wmma_f16(qa[c], b1, s1);
            }

            // ---- causal mask + online softmax (C layout) ----
            #pragma unroll
            for (int r = 0; r < 8; ++r) {
                const int row = i0 + r + 8 * half;
                const int c0  = j0 + m16;
                float v0 = s0[r] * scale;
                float v1 = s1[r] * scale;
                if (c0 > row)      v0 = -3e38f;
                if (c0 + 16 > row) v1 = -3e38f;

                const float bmax = grp16_max(fmaxf(v0, v1));
                const float mnew = fmaxf(mrow[r], bmax);
                const float alpha = __expf(mrow[r] - mnew);
                const float p0 = __expf(v0 - mnew);
                const float p1 = __expf(v1 - mnew);
                lrow[r] = lrow[r] * alpha + grp16_sum(p0 + p1);
                mrow[r] = mnew;
                #pragma unroll
                for (int t = 0; t < 16; ++t) o[t][r] *= alpha;
                s0[r] = p0;
                s1[r] = p1;
            }

            // ---- C-layout -> A-layout via per-wave LDS slice ----
            _Float16* pb = pbuf[wave];
            #pragma unroll
            for (int r = 0; r < 8; ++r) {
                const int pr = (r + 8 * half) * 32;
                pb[pr + m16]      = (_Float16)s0[r];
                pb[pr + 16 + m16] = (_Float16)s1[r];
            }
            const _Float16* prow = pb + m16 * 32;
            v8h plo = *(const v8h*)(prow + 8 * half);
            v8h phi = *(const v8h*)(prow + 16 + 8 * half);
            v16h pa = __builtin_shufflevector(plo, phi, 0,1,2,3,4,5,6,7,
                                                        8,9,10,11,12,13,14,15);

            // ---- P @ V from LDS: 16 tiles covering D=256 ----
            #pragma unroll
            for (int t = 0; t < 16; ++t) {
                const _Float16* vp = vtile + 16 * t + m16;
                v16h vb;
                #pragma unroll
                for (int i = 0; i < 8; ++i) {
                    vb[i]     = vp[(8 * half + i) * VT_PITCH];
                    vb[8 + i] = vp[(16 + 8 * half + i) * VT_PITCH];
                }
                o[t] = wmma_f16(pa, vb, o[t]);
            }
        }
        __syncthreads();   // protect tiles before next iteration's overwrite
    }

    // ---- epilogue: multiply by 1/l, write fp32 ----
    float rinv[8];
    #pragma unroll
    for (int r = 0; r < 8; ++r) rinv[r] = fast_rcp(lrow[r]);
    #pragma unroll
    for (int t = 0; t < 16; ++t) {
        #pragma unroll
        for (int r = 0; r < 8; ++r) {
            const int row = i0 + r + 8 * half;
            out[(size_t)row * D_DIM + 16 * t + m16] = o[t][r] * rinv[r];
        }
    }
}

extern "C" void kernel_launch(void* const* d_in, const int* in_sizes, int n_in,
                              void* d_out, int out_size, void* d_ws, size_t ws_size,
                              hipStream_t stream) {
    const float* x  = (const float*)d_in[0];
    const float* Wq = (const float*)d_in[1];
    const float* Wk = (const float*)d_in[2];
    const float* Wv = (const float*)d_in[3];

    _Float16* Qh = (_Float16*)d_ws;                       // [T,D]   2 MB
    _Float16* Kt = Qh + (size_t)T_DIM * D_DIM;            // [D,T]   2 MB
    _Float16* Vh = Kt + (size_t)T_DIM * D_DIM;            // [T,D]   2 MB
    _Float16* Wt = Vh + (size_t)T_DIM * D_DIM;            // [3,E,D] 3 MB

    const size_t wtotal = (size_t)3 * E_DIM * D_DIM;
    prep_w_kernel<<<(unsigned)((wtotal + 255) / 256), 256, 0, stream>>>(
        Wq, Wk, Wv, Wt);

    dim3 gproj(T_DIM / 16, D_DIM / 64, 3);
    proj_qkv_kernel<<<gproj, 32, 0, stream>>>(x, Wt, Qh, Kt, Vh);

    flash_attn_kernel<<<T_DIM / 128, 256, 0, stream>>>(
        Qh, Kt, Vh, (float*)d_out);
}
